// MultiHeadAttention_29781303230465
// MI455X (gfx1250) — matrix operand order repaired and checked by reading, not compiled, not run on hardware
//
#include <hip/hip_runtime.h>

// ---------------------------------------------------------------------------
// MI455X (gfx1250) multi-head attention, flash-attention style, bf16 WMMA.
//   B=8, S=2048, E=128, H=8, hs=16.  ~19.5 GFLOP; streaming softmax keeps the
//   512MB score tensor on-chip; all GEMMs use v_wmma_f32_16x16x32_bf16.
//   attn processes 2 query tiles per wave to reuse each K/V fragment load.
// ---------------------------------------------------------------------------

typedef __attribute__((ext_vector_type(16))) __bf16 v16bf;
typedef __attribute__((ext_vector_type(8)))  __bf16 v8bf;
typedef __attribute__((ext_vector_type(8)))  float  v8f;

#define B_  8
#define S_  2048
#define E_  128
#define H_  8
#define HS_ 16
#define NW  4   // waves per block (128 threads, wave32)

__device__ __forceinline__ v8f wmma_bf16(v16bf a, v16bf b, v8f c) {
  // 8-arg form: (neg_a, A, neg_b, B, c_mod, C, reuse_a, reuse_b)
  return __builtin_amdgcn_wmma_f32_16x16x32_bf16(false, a, false, b,
                                                 (short)0, c, false, false);
}

template <int BASE>
__device__ __forceinline__ void put4(v16bf& a, float4 u) {
  a[BASE + 0] = (__bf16)u.x; a[BASE + 1] = (__bf16)u.y;
  a[BASE + 2] = (__bf16)u.z; a[BASE + 3] = (__bf16)u.w;
}

// ---------------------------------------------------------------------------
// Kernel 0: weights fp32 -> bf16 (tiny; L2 resident afterwards)
// ---------------------------------------------------------------------------
__global__ void cvt_weights(const float* __restrict__ wq, const float* __restrict__ wk,
                            const float* __restrict__ wv, const float* __restrict__ wp,
                            __bf16* __restrict__ wqb, __bf16* __restrict__ wkb,
                            __bf16* __restrict__ wvb, __bf16* __restrict__ wpb, int n) {
  int i = blockIdx.x * blockDim.x + threadIdx.x;
  if (i < n) {
    wqb[i] = (__bf16)wq[i];
    wkb[i] = (__bf16)wk[i];
    wvb[i] = (__bf16)wv[i];
    wpb[i] = (__bf16)wp[i];
  }
}

// ---------------------------------------------------------------------------
// Kernel 1: QKV projection.  One wave = one 16-row s-tile, all 8 heads.
//   Q,K -> [B,H,S,16] row-major bf16;  V -> transposed [B,H,16,S] bf16
//   (so the P@V WMMA B-fragment is a contiguous 32B load per lane).
//   Per matrix: preload all 4 B-fragments (one clause), then 4 chained WMMAs.
// ---------------------------------------------------------------------------
__global__ void __launch_bounds__(NW * 32)
qkv_proj(const float* __restrict__ x,
         const __bf16* __restrict__ wqb, const __bf16* __restrict__ wkb,
         const __bf16* __restrict__ wvb,
         __bf16* __restrict__ qo, __bf16* __restrict__ ko, __bf16* __restrict__ vto) {
  __shared__ __align__(16) __bf16 lds[NW][16 * 16];
  const int lane  = threadIdx.x & 31;
  const int wave  = threadIdx.x >> 5;
  const int task  = blockIdx.x * NW + wave;      // B * S/16 = 1024 tasks
  const int stile = task & 127;
  const int b     = task >> 7;
  const int half  = lane >> 4;
  const int mrow  = lane & 15;
  const int s0    = stile * 16;
  __bf16* lp = &lds[wave][0];

  // A fragments of x-tile (16x128 fp32 -> 4 bf16 fragments of 16x32).
  // A-layout 16x32 (contiguous): elem j -> K = 16*half + j.
  const float* xp = x + ((size_t)b * S_ + s0 + mrow) * E_;
  v16bf a[4];
#pragma unroll
  for (int kt = 0; kt < 4; ++kt) {
    const float4* p0 = (const float4*)(xp + kt * 32 + 16 * half);
    const float4* p1 = (const float4*)(xp + kt * 32 + 16 * half + 8);
    put4<0>(a[kt], p0[0]); put4<4>(a[kt], p0[1]);
    put4<8>(a[kt], p1[0]); put4<12>(a[kt], p1[1]);
  }

  const size_t wrow = (size_t)mrow * E_;   // lane's d-row within a head
#pragma unroll 1
  for (int h = 0; h < H_; ++h) {
    const size_t woff = (size_t)h * HS_ * E_ + wrow + 16 * half;

    // ---- Q ----
    {
      v16bf bf[4];
#pragma unroll
      for (int kt = 0; kt < 4; ++kt)
        bf[kt] = *(const v16bf*)(wqb + woff + kt * 32);
      v8f c = {};
#pragma unroll
      for (int kt = 0; kt < 4; ++kt) c = wmma_bf16(a[kt], bf[kt], c);
      __bf16* outp = qo + (((size_t)b * H_ + h) * S_ + s0) * HS_;
#pragma unroll
      for (int r = 0; r < 8; ++r)
        outp[(size_t)(r + 8 * half) * HS_ + mrow] = (__bf16)c[r];
    }
    // ---- K ----
    {
      v16bf bf[4];
#pragma unroll
      for (int kt = 0; kt < 4; ++kt)
        bf[kt] = *(const v16bf*)(wkb + woff + kt * 32);
      v8f c = {};
#pragma unroll
      for (int kt = 0; kt < 4; ++kt) c = wmma_bf16(a[kt], bf[kt], c);
      __bf16* outp = ko + (((size_t)b * H_ + h) * S_ + s0) * HS_;
#pragma unroll
      for (int r = 0; r < 8; ++r)
        outp[(size_t)(r + 8 * half) * HS_ + mrow] = (__bf16)c[r];
    }
    // ---- V (transpose 16x16 tile through LDS, coalesced 16B stores) ----
    {
      v16bf bf[4];
#pragma unroll
      for (int kt = 0; kt < 4; ++kt)
        bf[kt] = *(const v16bf*)(wvb + woff + kt * 32);
      v8f c = {};
#pragma unroll
      for (int kt = 0; kt < 4; ++kt) c = wmma_bf16(a[kt], bf[kt], c);
#pragma unroll
      for (int r = 0; r < 8; ++r)
        lp[mrow * 16 + (r + 8 * half)] = (__bf16)c[r];   // lds[d][m]
      asm volatile("s_wait_dscnt 0x0" ::: "memory");
      const int d = lane >> 1, cc = (lane & 1) * 8;
      v8bf row = *(const v8bf*)(lp + d * 16 + cc);
      *(v8bf*)(vto + (((size_t)b * H_ + h) * HS_ + d) * S_ + s0 + cc) = row;
      asm volatile("s_wait_dscnt 0x0" ::: "memory");     // WAR guard
    }
  }
}

// ---------------------------------------------------------------------------
// Online-softmax update for one 16-query x 32-key score pair; stages P (bf16)
// into LDS in [row][key] order for the A-fragment re-load.
// ---------------------------------------------------------------------------
__device__ __forceinline__ void softmax_update(v8f s0v, v8f s1v,
                                               float (&mi)[8], float (&li)[8],
                                               v8f& o, __bf16* lp,
                                               int half, int mrow) {
#pragma unroll
  for (int r = 0; r < 8; ++r) {
    float x0 = s0v[r] * 4.0f, x1 = s1v[r] * 4.0f;   // reference *multiplies* sqrt(hs)=4
    float mx = fmaxf(x0, x1);
    mx = fmaxf(mx, __shfl_xor(mx, 1, 32));
    mx = fmaxf(mx, __shfl_xor(mx, 2, 32));
    mx = fmaxf(mx, __shfl_xor(mx, 4, 32));
    mx = fmaxf(mx, __shfl_xor(mx, 8, 32));          // masks 1/2/4/8 stay in 16-lane half
    float mn = fmaxf(mi[r], mx);
    float al = __expf(mi[r] - mn);
    float p0 = __expf(x0 - mn), p1 = __expf(x1 - mn);
    float rs = p0 + p1;
    rs += __shfl_xor(rs, 1, 32);
    rs += __shfl_xor(rs, 2, 32);
    rs += __shfl_xor(rs, 4, 32);
    rs += __shfl_xor(rs, 8, 32);
    li[r] = li[r] * al + rs;
    o[r]  = o[r] * al;
    mi[r] = mn;
    lp[(r + 8 * half) * 32 + mrow]      = (__bf16)p0;  // P[row][key 0..15]
    lp[(r + 8 * half) * 32 + 16 + mrow] = (__bf16)p1;  // P[row][key 16..31]
  }
}

// ---------------------------------------------------------------------------
// Kernel 2: flash attention.  One wave = one (b, h, 32-query block = 2 tiles).
//   Per 32-key step: 4 score WMMAs + 2 P@V WMMAs; every K/V fragment load is
//   reused by both query tiles (halves K/V read traffic vs 1 tile/wave).
// ---------------------------------------------------------------------------
__global__ void __launch_bounds__(NW * 32)
attn(const __bf16* __restrict__ qb, const __bf16* __restrict__ kb,
     const __bf16* __restrict__ vtb, __bf16* __restrict__ att) {
  __shared__ __align__(16) __bf16 lds[NW][2][16 * 32];
  const int lane = threadIdx.x & 31;
  const int wave = threadIdx.x >> 5;
  const int task = blockIdx.x * NW + wave;       // B*H*(S/32) = 4096 tasks
  const int qt2  = task & 63;                    // 32-query block index
  const int h    = (task >> 6) & 7;
  const int b    = task >> 9;
  const int half = lane >> 4;
  const int mrow = lane & 15;
  __bf16* lp0 = &lds[wave][0][0];
  __bf16* lp1 = &lds[wave][1][0];

  const __bf16* qp = qb  + (((size_t)b * H_ + h) * S_ + qt2 * 32) * HS_;
  const __bf16* kp = kb  + (((size_t)b * H_ + h) * S_) * HS_;
  const __bf16* vp = vtb + (((size_t)b * H_ + h) * HS_) * S_;

  // Q as A-fragments: elem j -> K = 16*half + j; half==0 holds hs 0..15, half==1 = pad 0.
  v16bf aq0 = {}, aq1 = {};
  if (half == 0) {
    aq0 = *(const v16bf*)(qp + mrow * HS_);
    aq1 = *(const v16bf*)(qp + (16 + mrow) * HS_);
  }

  float mi0[8], li0[8], mi1[8], li1[8];
#pragma unroll
  for (int r = 0; r < 8; ++r) {
    mi0[r] = -3.0e38f; li0[r] = 0.0f;
    mi1[r] = -3.0e38f; li1[r] = 0.0f;
  }
  v8f o0 = {}, o1 = {};

  for (int t0 = 0; t0 < S_; t0 += 32) {
    // K^T B-fragments: lane n = key (mrow); elem j -> k = 16*half + j.
    // half==1 lanes are the hs zero-pad region -> stay zero.
    v16bf bk0 = {}, bk1 = {};
    if (half == 0) {
      bk0 = *(const v16bf*)(kp + (size_t)(t0 + mrow) * HS_);
      bk1 = *(const v16bf*)(kp + (size_t)(t0 + 16 + mrow) * HS_);
    }
    // V B-fragment: lane n = d (mrow); elem j -> key 16*half + j.
    v16bf bv = *(const v16bf*)(vp + (size_t)mrow * S_ + t0 + 16 * half);

    v8f s00 = {}, s01 = {}, s10 = {}, s11 = {};
    s00 = wmma_bf16(aq0, bk0, s00);
    s01 = wmma_bf16(aq0, bk1, s01);
    s10 = wmma_bf16(aq1, bk0, s10);
    s11 = wmma_bf16(aq1, bk1, s11);

    if (t0 + 32 < S_) {  // pull next tile toward L2/L0 (global_prefetch_b8)
      __builtin_prefetch(kp + (size_t)(t0 + 32 + mrow) * HS_, 0, 0);
      __builtin_prefetch(vp + (size_t)mrow * S_ + t0 + 32, 0, 0);
    }

    softmax_update(s00, s01, mi0, li0, o0, lp0, half, mrow);
    softmax_update(s10, s11, mi1, li1, o1, lp1, half, mrow);
    asm volatile("s_wait_dscnt 0x0" ::: "memory");        // cross-lane RAW

    // P as A-fragments (16x32, all 32 K slots real), then P@V.
    v16bf ap0, ap1;
    {
      v8bf a0 = *(const v8bf*)(lp0 + mrow * 32 + 16 * half);
      v8bf a1 = *(const v8bf*)(lp0 + mrow * 32 + 16 * half + 8);
      v8bf a2 = *(const v8bf*)(lp1 + mrow * 32 + 16 * half);
      v8bf a3 = *(const v8bf*)(lp1 + mrow * 32 + 16 * half + 8);
#pragma unroll
      for (int j = 0; j < 8; ++j) {
        ap0[j] = a0[j]; ap0[8 + j] = a1[j];
        ap1[j] = a2[j]; ap1[8 + j] = a3[j];
      }
    }
    o0 = wmma_bf16(ap0, bv, o0);
    o1 = wmma_bf16(ap1, bv, o1);
    asm volatile("s_wait_dscnt 0x0" ::: "memory");        // WAR for next iter
  }

  // Normalize, write head slice of att[B][S][128] (cols h*16..h*16+15).
  __bf16* op0 = att + ((size_t)b * S_ + qt2 * 32) * E_ + h * HS_;
  __bf16* op1 = op0 + (size_t)16 * E_;
#pragma unroll
  for (int r = 0; r < 8; ++r) {
    op0[(size_t)(r + 8 * half) * E_ + mrow] = (__bf16)(o0[r] / li0[r]);
    op1[(size_t)(r + 8 * half) * E_ + mrow] = (__bf16)(o1[r] / li1[r]);
  }
}

// ---------------------------------------------------------------------------
// Kernel 3: output projection  y = att @ Wp^T + bp   (fp32 out).
//   Per N-tile: preload all 4 B-fragments, then 4 chained WMMAs.
// ---------------------------------------------------------------------------
__global__ void __launch_bounds__(NW * 32)
out_proj(const __bf16* __restrict__ att, const __bf16* __restrict__ wpb,
         const float* __restrict__ bp, float* __restrict__ y) {
  const int lane = threadIdx.x & 31;
  const int wave = threadIdx.x >> 5;
  const int task = blockIdx.x * NW + wave;       // B * S/16 = 1024 tasks
  const int st   = task & 127;
  const int b    = task >> 7;
  const int half = lane >> 4;
  const int mrow = lane & 15;

  const __bf16* ap_ = att + ((size_t)b * S_ + st * 16) * E_ + (size_t)mrow * E_;
  v16bf a[4];
#pragma unroll
  for (int kt = 0; kt < 4; ++kt) {
    v8bf lo = *(const v8bf*)(ap_ + kt * 32 + 16 * half);
    v8bf hi = *(const v8bf*)(ap_ + kt * 32 + 16 * half + 8);
#pragma unroll
    for (int j = 0; j < 8; ++j) { a[kt][j] = lo[j]; a[kt][8 + j] = hi[j]; }
  }

  float* yp = y + ((size_t)b * S_ + st * 16) * E_;
#pragma unroll 1
  for (int nt = 0; nt < 8; ++nt) {
    // B[k=c][n=e] = Wp[e][c]: lane e = nt*16+mrow; elem j -> c = kt*32+16*half+j
    const __bf16* wrow = wpb + (size_t)(nt * 16 + mrow) * E_ + 16 * half;
    v16bf bw[4];
#pragma unroll
    for (int kt = 0; kt < 4; ++kt)
      bw[kt] = *(const v16bf*)(wrow + kt * 32);
    v8f c = {};
#pragma unroll
    for (int kt = 0; kt < 4; ++kt) c = wmma_bf16(a[kt], bw[kt], c);
    float bias = bp[nt * 16 + mrow];
#pragma unroll
    for (int r = 0; r < 8; ++r)   // lanes 0-15 per reg = 64B contiguous stores
      yp[(size_t)(r + 8 * half) * E_ + nt * 16 + mrow] = c[r] + bias;
  }
}

// ---------------------------------------------------------------------------
extern "C" void kernel_launch(void* const* d_in, const int* in_sizes, int n_in,
                              void* d_out, int out_size, void* d_ws, size_t ws_size,
                              hipStream_t stream) {
  (void)in_sizes; (void)n_in; (void)out_size; (void)ws_size;
  // setup_inputs order: x, Wk, Wq, Wv, Wp, bp
  const float* x  = (const float*)d_in[0];
  const float* Wk = (const float*)d_in[1];
  const float* Wq = (const float*)d_in[2];
  const float* Wv = (const float*)d_in[3];
  const float* Wp = (const float*)d_in[4];
  const float* bp = (const float*)d_in[5];
  float* y = (float*)d_out;

  // Workspace layout (bf16 elements): weights, Q, K, Vt, att. ~17 MB total.
  __bf16* ws  = (__bf16*)d_ws;
  const size_t WSZ  = (size_t)H_ * HS_ * E_;          // 16384
  const size_t QKV  = (size_t)B_ * H_ * S_ * HS_;     // 2,097,152
  __bf16* wqb = ws;
  __bf16* wkb = wqb + WSZ;
  __bf16* wvb = wkb + WSZ;
  __bf16* wpb = wvb + WSZ;
  __bf16* qbf = wpb + WSZ;
  __bf16* kbf = qbf + QKV;
  __bf16* vtb = kbf + QKV;
  __bf16* atb = vtb + QKV;                            // [B][S][128] bf16

  cvt_weights<<<(int)((WSZ + 255) / 256), 256, 0, stream>>>(
      Wq, Wk, Wv, Wp, wqb, wkb, wvb, wpb, (int)WSZ);

  // B * S/16 / NW = 256 blocks
  qkv_proj<<<(B_ * (S_ / 16)) / NW, NW * 32, 0, stream>>>(
      x, wqb, wkb, wvb, qbf, kbf, vtb);

  // B * H * (S/32) / NW = 1024 blocks
  attn<<<(B_ * H_ * (S_ / 32)) / NW, NW * 32, 0, stream>>>(qbf, kbf, vtb, atb);

  out_proj<<<(B_ * (S_ / 16)) / NW, NW * 32, 0, stream>>>(atb, wpb, bp, y);
}
